// UpdateCoor_88545045775094
// MI455X (gfx1250) — matrix operand order repaired and checked by reading, not compiled, not run on hardware
//
#include <hip/hip_runtime.h>
#include <hip/hip_bf16.h>

typedef __attribute__((ext_vector_type(16))) __bf16 v16bf;
typedef __attribute__((ext_vector_type(8)))  float  v8f;

#define NN 256
#define HH 8
#define NEG_MAX (-3.402823466e+38f)

__device__ __forceinline__ __bf16 bfc(float x){ return (__bf16)x; }

__device__ __forceinline__ v8f wmma32(v16bf a, v16bf b, v8f c){
  return __builtin_amdgcn_wmma_f32_16x16x32_bf16(false, a, false, b, (short)0, c, false, false);
}

// CDNA5 async global->LDS bulk copy (ASYNCcnt-tracked, no VGPR round-trip).
__device__ __forceinline__ void async_cp_b128(unsigned lds_off, unsigned long long gaddr){
  asm volatile("global_load_async_to_lds_b128 %0, %1, off" :: "v"(lds_off), "v"(gaddr) : "memory");
}
__device__ __forceinline__ void wait_async0(){
  asm volatile("s_wait_asynccnt 0x0" ::: "memory");
}

// A: row-major MxK bf16 matrix (global or LDS), fragment for (mBase,kBase) tile.
// CDNA5 16-bit A layout: lanes 0-15 = rows, hold K {0..7,16..23}; lanes 16-31 hold K {8..15,24..31}.
__device__ __forceinline__ v16bf load_a(const __bf16* __restrict__ A, int lda,
                                        int mBase, int kBase, int lane){
  const __bf16* p = A + (mBase + (lane & 15)) * lda + kBase + ((lane & 16) ? 8 : 0);
  v16bf r;
#pragma unroll
  for (int e = 0; e < 8; ++e){ r[e] = p[e]; r[e + 8] = p[e + 16]; }
  return r;
}

// B: row-major KxN bf16 matrix; lanes 0-15 = col n hold K {0..7,16..23}, lanes 16-31 hold K {8..15,24..31}.
__device__ __forceinline__ v16bf load_b(const __bf16* __restrict__ Bm, int ldb,
                                        int kBase, int nBase, int lane){
  const __bf16* p = Bm + (kBase + ((lane & 16) ? 8 : 0)) * ldb + nBase + (lane & 15);
  v16bf r;
#pragma unroll
  for (int e = 0; e < 8; ++e){ r[e] = p[e * ldb]; r[e + 8] = p[(e + 16) * ldb]; }
  return r;
}

// ---------------------------------------------------------------- prep: fp32 -> bf16 weights
__global__ void prep_kernel(
    const float* __restrict__ eWv1, const float* __restrict__ eWsca,
    const float* __restrict__ eWv2, const float* __restrict__ eWgate,
    const float* __restrict__ aW1, const float* __restrict__ ab1, const float* __restrict__ aW2,
    __bf16* __restrict__ Wv1b, __bf16* __restrict__ Wscab,
    __bf16* __restrict__ Wv2b, __bf16* __restrict__ Wgateb,
    __bf16* __restrict__ W1p, float* __restrict__ b1p, float* __restrict__ W2p)
{
  int idx = blockIdx.x * blockDim.x + threadIdx.x;
  if      (idx < 9216)  Wv1b[idx] = bfc(eWv1[idx]);
  else if (idx < 46080) Wscab[idx - 9216]  = bfc(eWsca[idx - 9216]);
  else if (idx < 52224) Wv2b[idx - 46080]  = bfc(eWv2[idx - 46080]);
  else if (idx < 60416) Wgateb[idx - 52224] = bfc(eWgate[idx - 52224]);
  else if (idx < 61440){ int k = idx - 60416; int m = k >> 5, kk = k & 31;
                         W1p[k] = (m < 24 && kk < 24) ? bfc(aW1[m * 24 + kk]) : bfc(0.f); }
  else if (idx < 61472){ int m = idx - 61440; b1p[m] = (m < 24) ? ab1[m] : 0.f; }
  else if (idx < 61504){ int m = idx - 61472; W2p[m] = (m < 24) ? aW2[m] : 0.f; }
}

// ---------------------------------------------------------------- node qkv GVL (VALU, tiny)
__global__ __launch_bounds__(256) void qkv_kernel(
    const float* __restrict__ x_sca, const float* __restrict__ x_vec,
    const float* __restrict__ Wgate, const float* __restrict__ Wsca,
    const float* __restrict__ Wv1,   const float* __restrict__ Wv2,
    const float* __restrict__ bgate, const float* __restrict__ bsca,
    float* __restrict__ xq_s, float* __restrict__ xk_s, float* __restrict__ xc_s,
    float* __restrict__ xq_v, float* __restrict__ xk_v, float* __restrict__ xc_v)
{
  __shared__ float xv[192];
  __shared__ float v1[576];
  __shared__ float cat[320];
  __shared__ float so[384];
  __shared__ float gl[192];
  const int t = threadIdx.x;
  const long node = blockIdx.x;
  if (t < 192) xv[t] = x_vec[node * 192 + t];
  if (t < 128) cat[192 + t] = x_sca[node * 128 + t];
  __syncthreads();
  for (int idx = t; idx < 576; idx += 256){
    int d = idx / 3, c = idx % 3;
    float a = 0.f;
    for (int e = 0; e < 64; ++e) a += Wv1[d * 64 + e] * xv[e * 3 + c];
    v1[idx] = a;
  }
  __syncthreads();
  if (t < 192) cat[t] = sqrtf(v1[t*3]*v1[t*3] + v1[t*3+1]*v1[t*3+1] + v1[t*3+2]*v1[t*3+2]);
  __syncthreads();
  for (int o = t; o < 384; o += 256){
    float a = bsca[o];
    for (int k = 0; k < 320; ++k) a += Wsca[o * 320 + k] * cat[k];
    so[o] = a;
    if      (o < 128) xq_s[node * 128 + o]       = a;
    else if (o < 256) xk_s[node * 128 + o - 128] = a;
    else              xc_s[node * 128 + o - 256] = a;
  }
  __syncthreads();
  for (int d = t; d < 192; d += 256){
    float a = bgate[d];
    for (int o = 0; o < 384; ++o) a += Wgate[d * 384 + o] * so[o];
    gl[d] = 1.f / (1.f + __expf(-a));
  }
  __syncthreads();
  for (int idx = t; idx < 576; idx += 256){
    int d = idx / 3, c = idx % 3;
    float a = 0.f;
    for (int e = 0; e < 192; ++e) a += Wv2[d * 192 + e] * v1[e * 3 + c];
    a *= gl[d];
    if      (d < 64)  xq_v[node * 192 + d * 3 + c]        = a;
    else if (d < 128) xk_v[node * 192 + (d - 64) * 3 + c] = a;
    else              xc_v[node * 192 + (d - 128) * 3 + c] = a;
  }
}

// ---------------------------------------------------------------- edge GVL + attention (WMMA)
// One WG (128 thr = 4 waves) per (b, i, 16-wide j tile).
// LDS layout (aliased across stages, ~50 KB):
#define SM_BUF0   0        // bf16[96*48]  ev -> later v1 (bf16)
#define SM_BUF1   9216     // f32 [96*48]  v1 (f32)
#define SM_SCAB   9216     //   later: bf16[128*16] sca (bf16)
#define SM_V2F    13312    //   later: f32 [64*48]  v2
#define SM_BUF2   27648    // bf16[288*16] s  -> later bf16[32*128] feat
#define SM_SCAF   36864    // f32 [16*128] sca_out (+bias); stage0: async staging (10 KB)
#define SM_GF     45056    // f32 [16*64]  gate
#define SM_XQS    49152    // f32 [128]
#define SM_XQV    49664    // f32 [192]
#define SM_REL    50432    // f32 [48]
#define SM_DIST   50624    // f32 [16]
#define SM_TOTAL  50688

__global__ __launch_bounds__(128) void edge_kernel(
    const float* __restrict__ edge_sca, const float* __restrict__ edge_vec,
    const unsigned char* __restrict__ mask, const float* __restrict__ coor,
    const float* __restrict__ W_vecexp, const float* __restrict__ dist_scale,
    const float* __restrict__ eb_bsca, const float* __restrict__ eb_bgate,
    const __bf16* __restrict__ Wv1b, const __bf16* __restrict__ Wscab,
    const __bf16* __restrict__ Wv2b, const __bf16* __restrict__ Wgateb,
    const __bf16* __restrict__ W1p, const float* __restrict__ b1p,
    const float* __restrict__ W2p, const float* __restrict__ attb2,
    const float* __restrict__ xq_s, const float* __restrict__ xk_s,
    const float* __restrict__ xq_v, const float* __restrict__ xk_v,
    float* __restrict__ logit_s, float* __restrict__ logit_v,
    float* __restrict__ attc)
{
  __shared__ __align__(16) char smem[SM_TOTAL];
  __bf16* ev_b   = (__bf16*)(smem + SM_BUF0);
  __bf16* v1_b   = (__bf16*)(smem + SM_BUF0);
  float*  v1_f   = (float*) (smem + SM_BUF1);
  __bf16* sca_b  = (__bf16*)(smem + SM_SCAB);
  float*  v2_f   = (float*) (smem + SM_V2F);
  __bf16* s_b    = (__bf16*)(smem + SM_BUF2);
  __bf16* feat_b = (__bf16*)(smem + SM_BUF2);
  float*  sca_f  = (float*) (smem + SM_SCAF);
  float*  g_f    = (float*) (smem + SM_GF);
  float*  xqs_l  = (float*) (smem + SM_XQS);
  float*  xqv_l  = (float*) (smem + SM_XQV);
  float*  rel_l  = (float*) (smem + SM_REL);
  float*  dist_l = (float*) (smem + SM_DIST);
  // async staging buffers alias sca_f/g_f (dead until G2)
  float*  stage_ev = (float*)(smem + SM_SCAF);         // 16*96 f32 = 6144 B
  float*  stage_es = (float*)(smem + SM_SCAF + 6144);  // 16*64 f32 = 4096 B

  const int tid  = threadIdx.x;
  const int wave = tid >> 5, lane = tid & 31;
  const int j0   = (blockIdx.x & 15) << 4;
  const int i    = (blockIdx.x >> 4) & 255;
  const int b    = blockIdx.x >> 12;
  const long nodeI = (long)b * NN + i;

  // ---- stage 0: kick off async global->LDS staging of this tile's edge features
  {
    const float* gev = edge_vec + (((long)b * NN + i) * NN + j0) * 96;   // 6144 B contiguous
    const float* ges = edge_sca + (((long)b * NN + i) * NN + j0) * 64;   // 4096 B contiguous
    unsigned lds_ev = (unsigned)(uintptr_t)stage_ev;
    unsigned lds_es = (unsigned)(uintptr_t)stage_es;
#pragma unroll
    for (int q = 0; q < 3; ++q){
      int chunk = q * 128 + tid;                       // 384 x 16B chunks
      async_cp_b128(lds_ev + chunk * 16, (unsigned long long)(uintptr_t)((const char*)gev + chunk * 16));
    }
#pragma unroll
    for (int q = 0; q < 2; ++q){
      int chunk = q * 128 + tid;                       // 256 x 16B chunks
      async_cp_b128(lds_es + chunk * 16, (unsigned long long)(uintptr_t)((const char*)ges + chunk * 16));
    }
    __builtin_prefetch(xk_s + ((long)b * NN + j0) * 128, 0, 3);
    __builtin_prefetch(xk_v + ((long)b * NN + j0) * 192, 0, 3);
  }

  // ---- stage 0a (overlaps async copies): rel/dist + node-i features
  if (tid < 16){
    int j = j0 + tid;
    float r0 = coor[nodeI*3]   - coor[((long)b*NN + j)*3];
    float r1 = coor[nodeI*3+1] - coor[((long)b*NN + j)*3+1];
    float r2 = coor[nodeI*3+2] - coor[((long)b*NN + j)*3+2];
    rel_l[tid*3] = r0; rel_l[tid*3+1] = r1; rel_l[tid*3+2] = r2;
    dist_l[tid] = sqrtf(r0*r0 + r1*r1 + r2*r2);
  }
  if (tid < 128) xqs_l[tid] = xq_s[nodeI * 128 + tid];
  for (int idx = tid; idx < 192; idx += 128) xqv_l[idx] = xq_v[nodeI * 192 + idx];
  wait_async0();
  __syncthreads();

  // ---- stage 0b: ev (96 x 48) and s rows 96..287 from staged LDS (f32 -> bf16)
  for (int idx = tid; idx < 96 * 48; idx += 128){
    int k = idx / 48, col = idx % 48, e = col / 3, c = col % 3;
    float v;
    if (k < 32) v = stage_ev[e * 96 + k * 3 + c];
    else        v = W_vecexp[k - 32] * rel_l[e * 3 + c];
    ev_b[k * 48 + col] = bfc(v);
  }
  const float ds0 = dist_scale[0];
  for (int idx = tid; idx < 192 * 16; idx += 128){
    int r = idx >> 4, e = idx & 15;
    float v;
    if (r < 64)       v = stage_es[e * 64 + r];
    else if (r == 64) v = dist_l[e] * ds0;
    else { int g = r - 65;
           float enc = fminf(dist_l[e], 2.0f) - (float)g * (1.0f / 63.0f);
           v = __expf(-1984.5f * enc * enc); }
    s_b[(96 + r) * 16 + e] = bfc(v);
  }
  __syncthreads();

  // ---- G1: v1 = W_vec1(96x96) @ ev(96x48)
  for (int t = wave; t < 18; t += 4){
    int mt = t / 3, nt = t % 3;
    v8f c = {};
#pragma unroll
    for (int kt = 0; kt < 3; ++kt)
      c = wmma32(load_a(Wv1b, 96, mt*16, kt*32, lane), load_b(ev_b, 48, kt*32, nt*16, lane), c);
    int n  = nt * 16 + (lane & 15);
    int m0 = mt * 16 + ((lane & 16) ? 8 : 0);
#pragma unroll
    for (int r = 0; r < 8; ++r) v1_f[(m0 + r) * 48 + n] = c[r];
  }
  __syncthreads();

  // ---- stage 2: v1 norms -> s rows 0..95 ; v1 -> bf16 (aliases old ev buffer)
  for (int idx = tid; idx < 96 * 16; idx += 128){
    int d = idx >> 4, e = idx & 15;
    float a0 = v1_f[d*48 + e*3], a1 = v1_f[d*48 + e*3 + 1], a2 = v1_f[d*48 + e*3 + 2];
    s_b[d * 16 + e] = bfc(sqrtf(a0*a0 + a1*a1 + a2*a2));
  }
  for (int idx = tid; idx < 96 * 48; idx += 128) v1_b[idx] = bfc(v1_f[idx]);
  __syncthreads();

  // ---- G2: sca_out = W_sca(128x288) @ s(288x16)  (+bias)
  for (int t = wave; t < 8; t += 4){
    v8f c = {};
#pragma unroll
    for (int kt = 0; kt < 9; ++kt)
      c = wmma32(load_a(Wscab, 288, t*16, kt*32, lane), load_b(s_b, 16, kt*32, 0, lane), c);
    int n  = lane & 15;
    int m0 = t * 16 + ((lane & 16) ? 8 : 0);
#pragma unroll
    for (int r = 0; r < 8; ++r){ int m = m0 + r; sca_f[n * 128 + m] = c[r] + eb_bsca[m]; }
  }
  __syncthreads();
  for (int idx = tid; idx < 128 * 16; idx += 128){
    int m = idx >> 4, e = idx & 15;
    sca_b[m * 16 + e] = bfc(sca_f[e * 128 + m]);
  }
  __syncthreads();

  // ---- G3: v2 = W_vec2(64x96) @ v1 ; G4: gate = sigmoid(W_gate(64x128) @ sca + b)
  for (int t = wave; t < 16; t += 4){
    if (t < 12){
      int mt = t / 3, nt = t % 3;
      v8f c = {};
#pragma unroll
      for (int kt = 0; kt < 3; ++kt)
        c = wmma32(load_a(Wv2b, 96, mt*16, kt*32, lane), load_b(v1_b, 48, kt*32, nt*16, lane), c);
      int n  = nt * 16 + (lane & 15);
      int m0 = mt * 16 + ((lane & 16) ? 8 : 0);
#pragma unroll
      for (int r = 0; r < 8; ++r) v2_f[(m0 + r) * 48 + n] = c[r];
    } else {
      int mt = t - 12;
      v8f c = {};
#pragma unroll
      for (int kt = 0; kt < 4; ++kt)
        c = wmma32(load_a(Wgateb, 128, mt*16, kt*32, lane), load_b(sca_b, 16, kt*32, 0, lane), c);
      int n  = lane & 15;
      int m0 = mt * 16 + ((lane & 16) ? 8 : 0);
#pragma unroll
      for (int r = 0; r < 8; ++r){ int m = m0 + r;
        g_f[n * 64 + m] = 1.0f / (1.0f + __expf(-(c[r] + eb_bgate[m]))); }
    }
  }
  __syncthreads();

  // ---- attention products + logits + feat assembly (feat aliases s buffer)
  {
    const int e = tid & 15, h = tid >> 4;
    const int j = j0 + e;
    const long nodeJ = (long)b * NN + j;
    const bool mj = mask[nodeJ] != 0;
    const float* xks = xk_s + nodeJ * 128;
    const float* xkv = xk_v + nodeJ * 192;
    float ssum = 0.f;
#pragma unroll
    for (int s = 0; s < 16; ++s){
      int ch = h * 16 + s;
      float r = xqs_l[ch] * xks[ch] * sca_f[e * 128 + ch];
      feat_b[s * 128 + tid] = bfc(r);
      ssum += r;
    }
    float vsum = 0.f;
#pragma unroll
    for (int d = 0; d < 8; ++d){
      int hd = h * 8 + d;
      float g = g_f[e * 64 + hd];
      float nr = 0.f;
#pragma unroll
      for (int c = 0; c < 3; ++c){
        float rv = xqv_l[hd*3 + c] * xkv[hd*3 + c] * (g * v2_f[hd * 48 + e * 3 + c]);
        vsum += rv; nr += rv * rv;
      }
      feat_b[(16 + d) * 128 + tid] = bfc(sqrtf(nr));
    }
#pragma unroll
    for (int r = 24; r < 32; ++r) feat_b[r * 128 + tid] = bfc(0.f);
    long eidx = (((long)b * NN + i) * NN + j) * HH + h;
    logit_s[eidx] = mj ? ssum * 0.25f                 : NEG_MAX;
    logit_v[eidx] = mj ? vsum * 0.2041241452319315f   : NEG_MAX;  // 1/sqrt(24)
  }
  __syncthreads();

  // ---- G5: att MLP  h1 = leaky(W1p(32x32) @ feat(32x128) + b1) ; att_c = W2.h1 + b2
  {
    const float b2 = attb2[0];
    for (int nt = wave * 2; nt < wave * 2 + 2; ++nt){
      float part = 0.f;
#pragma unroll
      for (int mt = 0; mt < 2; ++mt){
        v8f c = {};
        c = wmma32(load_a(W1p, 32, mt*16, 0, lane), load_b(feat_b, 128, 0, nt*16, lane), c);
        int m0 = mt * 16 + ((lane & 16) ? 8 : 0);
#pragma unroll
        for (int r = 0; r < 8; ++r){
          int m = m0 + r;
          float h1 = c[r] + b1p[m];
          h1 = (h1 > 0.f) ? h1 : 0.01f * h1;
          part += W2p[m] * h1;
        }
      }
      part += __shfl_xor(part, 16, 32);
      if (lane < 16){
        int col = nt * 16 + lane;
        int e2 = col & 15, h2 = col >> 4;
        int j2 = j0 + e2;
        float mj = mask[(long)b * NN + j2] ? 1.f : 0.f;
        long eidx = (((long)b * NN + i) * NN + j2) * HH + h2;
        attc[eidx] = (part + b2) * mj;
      }
    }
  }
}

// ---------------------------------------------------------------- softmax + aggregation
__global__ __launch_bounds__(256) void agg_kernel(
    const float* __restrict__ logit_s, const float* __restrict__ logit_v,
    const float* __restrict__ attc, const float* __restrict__ xc_s,
    const float* __restrict__ xc_v, const float* __restrict__ coor,
    const float* __restrict__ cn_scale,
    float* __restrict__ sc_o, float* __restrict__ vc_o, float* __restrict__ atc_o)
{
  __shared__ float ps[8 * 256];
  __shared__ float pv[8 * 256];
  __shared__ float ac[8 * 256];
  const int t = threadIdx.x;
  const int node = blockIdx.x;
  const int b = node >> 8;
  {
    long base = ((long)node * NN + t) * HH;
    for (int h = 0; h < 8; ++h){
      ps[h * 256 + t] = logit_s[base + h];
      pv[h * 256 + t] = logit_v[base + h];
      ac[h * 256 + t] = attc[base + h];
    }
  }
  __syncthreads();
  // per-head softmax over j (wave w owns head w; 32 lanes stride the 256 j's)
  const int h = t >> 5, l = t & 31;
  float* arr0 = ps; float* arr1 = pv;
  for (int a2 = 0; a2 < 2; ++a2){
    float* X = (a2 == 0 ? arr0 : arr1) + h * 256;
    float m = NEG_MAX;
    for (int j = l; j < 256; j += 32) m = fmaxf(m, X[j]);
    for (int off = 16; off > 0; off >>= 1) m = fmaxf(m, __shfl_xor(m, off, 32));
    float s = 0.f;
    for (int j = l; j < 256; j += 32){ float e = __expf(X[j] - m); X[j] = e; s += e; }
    for (int off = 16; off > 0; off >>= 1) s += __shfl_xor(s, off, 32);
    float inv = 1.f / s;
    for (int j = l; j < 256; j += 32) X[j] *= inv;
  }
  __syncthreads();
  const float cn = cn_scale[0];
  if (t < 128){
    int hh = t >> 4, s = t & 15;
    float a = 0.f;
    for (int j = 0; j < 256; ++j) a += ps[hh * 256 + j] * xc_s[((long)b * NN + j) * 128 + hh * 16 + s];
    sc_o[(long)node * 128 + hh * 16 + s] = a;
  }
  if (t < 192){
    int hh = t / 24, r = t % 24;
    float a = 0.f;
    for (int j = 0; j < 256; ++j) a += pv[hh * 256 + j] * xc_v[((long)b * NN + j) * 192 + hh * 24 + r];
    vc_o[(long)node * 192 + hh * 24 + r] = a;
  }
  if (t < 24){
    int hh = t / 3, c = t % 3;
    float x0 = coor[(long)node * 3], x1 = coor[(long)node * 3 + 1], x2 = coor[(long)node * 3 + 2];
    float a = 0.f;
    for (int j = 0; j < 256; ++j){
      long nj = (long)b * NN + j;
      float r0 = x0 - coor[nj * 3], r1 = x1 - coor[nj * 3 + 1], r2 = x2 - coor[nj * 3 + 2];
      float rn = sqrtf(r0 * r0 + r1 * r1 + r2 * r2);
      if (rn == 0.f) rn = 1e8f;
      rn = fmaxf(rn, 1e-8f);
      float rc = (c == 0) ? r0 : ((c == 1) ? r1 : r2);
      a += ac[hh * 256 + j] * (rc / rn) * cn;
    }
    atc_o[(long)node * 24 + hh * 3 + c] = a;
  }
}

// ---------------------------------------------------------------- per-node tail GVPs
__global__ __launch_bounds__(32) void final_kernel(
    const float* __restrict__ sc_i, const float* __restrict__ vc_i,
    const float* __restrict__ atc_i, const float* __restrict__ coor,
    const float* __restrict__ pW_act,
    const float* __restrict__ pg_Wgate, const float* __restrict__ pg_Wsca,
    const float* __restrict__ pg_Wv1,   const float* __restrict__ pg_Wv2,
    const float* __restrict__ pg_bgate, const float* __restrict__ pg_bsca,
    const float* __restrict__ pl_Wgate, const float* __restrict__ pl_Wsca,
    const float* __restrict__ pl_Wv1,   const float* __restrict__ pl_Wv2,
    const float* __restrict__ pl_bgate, const float* __restrict__ pl_bsca,
    const float* __restrict__ combine, float* __restrict__ out)
{
  __shared__ float contrib[8][3];
  const long node = blockIdx.x;
  const int t = threadIdx.x;
  if (t < 8){
    const int h = t;
    float sc[16], vec[8][3];
    for (int s = 0; s < 16; ++s) sc[s] = sc_i[node * 128 + h * 16 + s];
    for (int d = 0; d < 8; ++d)
      for (int c = 0; c < 3; ++c) vec[d][c] = vc_i[node * 192 + h * 24 + d * 3 + c];
    // coor_gvp.gvl (16,8 -> 16,8)
    float v1[8][3];
    for (int d = 0; d < 8; ++d) for (int c = 0; c < 3; ++c){
      float a = 0.f; for (int e = 0; e < 8; ++e) a += pg_Wv1[d*8+e] * vec[e][c]; v1[d][c] = a; }
    float cat[24];
    for (int d = 0; d < 8; ++d)
      cat[d] = sqrtf(v1[d][0]*v1[d][0] + v1[d][1]*v1[d][1] + v1[d][2]*v1[d][2]);
    for (int s = 0; s < 16; ++s) cat[8 + s] = sc[s];
    float so[16];
    for (int o = 0; o < 16; ++o){ float a = pg_bsca[o];
      for (int k = 0; k < 24; ++k) a += pg_Wsca[o*24+k] * cat[k]; so[o] = a; }
    float v2g[8][3];
    for (int d = 0; d < 8; ++d) for (int c = 0; c < 3; ++c){
      float a = 0.f; for (int e = 0; e < 8; ++e) a += pg_Wv2[d*8+e] * v1[e][c]; v2g[d][c] = a; }
    float g8[8];
    for (int d = 0; d < 8; ++d){ float a = pg_bgate[d];
      for (int o = 0; o < 16; ++o) a += pg_Wgate[d*16+o] * so[o];
      g8[d] = 1.f / (1.f + __expf(-a)); }
    for (int d = 0; d < 8; ++d) for (int c = 0; c < 3; ++c) vec[d][c] = g8[d] * v2g[d][c];
    float sca[16];
    for (int o = 0; o < 16; ++o){ float v = so[o]; sca[o] = v > 0.f ? v : 0.01f * v; }
    // vnl_lrelu(W_act)
    float vo[8][3];
    for (int d = 0; d < 8; ++d) for (int c = 0; c < 3; ++c){
      float a = 0.f; for (int e = 0; e < 8; ++e) a += pW_act[d*8+e] * vec[e][c]; vo[d][c] = a; }
    for (int d = 0; d < 8; ++d){
      float dot = vec[d][0]*vo[d][0] + vec[d][1]*vo[d][1] + vec[d][2]*vo[d][2];
      float nsq = vo[d][0]*vo[d][0] + vo[d][1]*vo[d][1] + vo[d][2]*vo[d][2];
      if (dot < 0.f){
        float f = dot / (nsq + 1e-8f);
        for (int c = 0; c < 3; ++c) vec[d][c] = 0.01f * vec[d][c] + 0.99f * (vec[d][c] - vo[d][c] * f);
      }
    }
    // coor_gvl (16,8 -> 16,1), only vec output used
    float v1b[8][3];
    for (int d = 0; d < 8; ++d) for (int c = 0; c < 3; ++c){
      float a = 0.f; for (int e = 0; e < 8; ++e) a += pl_Wv1[d*8+e] * vec[e][c]; v1b[d][c] = a; }
    float cat2[24];
    for (int d = 0; d < 8; ++d)
      cat2[d] = sqrtf(v1b[d][0]*v1b[d][0] + v1b[d][1]*v1b[d][1] + v1b[d][2]*v1b[d][2]);
    for (int s = 0; s < 16; ++s) cat2[8 + s] = sca[s];
    float so2[16];
    for (int o = 0; o < 16; ++o){ float a = pl_bsca[o];
      for (int k = 0; k < 24; ++k) a += pl_Wsca[o*24+k] * cat2[k]; so2[o] = a; }
    float v2l[3];
    for (int c = 0; c < 3; ++c){ float a = 0.f;
      for (int e = 0; e < 8; ++e) a += pl_Wv2[e] * v1b[e][c]; v2l[c] = a; }
    float ga = pl_bgate[0];
    for (int o = 0; o < 16; ++o) ga += pl_Wgate[o] * so2[o];
    float gg = 1.f / (1.f + __expf(-ga));
    float chx = combine[h], cax = combine[8 + h];
    for (int c = 0; c < 3; ++c)
      contrib[h][c] = chx * (gg * v2l[c]) + cax * atc_i[node * 24 + h * 3 + c];
  }
  __syncthreads();
  if (t < 3){
    float a = 0.f;
    for (int hh = 0; hh < 8; ++hh) a += contrib[hh][t];
    out[node * 3 + t] = coor[node * 3 + t] + a;
  }
}

// ---------------------------------------------------------------- launch
extern "C" void kernel_launch(void* const* d_in, const int* in_sizes, int n_in,
                              void* d_out, int out_size, void* d_ws, size_t ws_size,
                              hipStream_t stream)
{
  (void)in_sizes; (void)n_in; (void)out_size; (void)ws_size;
  const float* x_sca    = (const float*)d_in[0];
  const float* x_vec    = (const float*)d_in[1];
  const float* edge_sca = (const float*)d_in[2];
  const float* edge_vec = (const float*)d_in[3];
  const unsigned char* mask = (const unsigned char*)d_in[4];
  const float* coor     = (const float*)d_in[5];
  // params, jax pytree (sorted dict keys) flatten order
  const float* W_vecexp   = (const float*)d_in[6];
  const float* att_W1     = (const float*)d_in[7];
  const float* att_W2     = (const float*)d_in[8];
  const float* att_b1     = (const float*)d_in[9];
  const float* att_b2     = (const float*)d_in[10];
  const float* cn_scale   = (const float*)d_in[11];
  const float* combine    = (const float*)d_in[12];
  const float* cl_Wgate   = (const float*)d_in[13];
  const float* cl_Wsca    = (const float*)d_in[14];
  const float* cl_Wv1     = (const float*)d_in[15];
  const float* cl_Wv2     = (const float*)d_in[16];
  const float* cl_bgate   = (const float*)d_in[17];
  const float* cl_bsca    = (const float*)d_in[18];
  const float* cp_Wact    = (const float*)d_in[19];
  const float* cp_Wgate   = (const float*)d_in[20];
  const float* cp_Wsca    = (const float*)d_in[21];
  const float* cp_Wv1     = (const float*)d_in[22];
  const float* cp_Wv2     = (const float*)d_in[23];
  const float* cp_bgate   = (const float*)d_in[24];
  const float* cp_bsca    = (const float*)d_in[25];
  const float* dist_scale = (const float*)d_in[26];
  const float* eb_Wgate   = (const float*)d_in[27];
  const float* eb_Wsca    = (const float*)d_in[28];
  const float* eb_Wv1     = (const float*)d_in[29];
  const float* eb_Wv2     = (const float*)d_in[30];
  const float* eb_bgate   = (const float*)d_in[31];
  const float* eb_bsca    = (const float*)d_in[32];
  const float* qk_Wgate   = (const float*)d_in[33];
  const float* qk_Wsca    = (const float*)d_in[34];
  const float* qk_Wv1     = (const float*)d_in[35];
  const float* qk_Wv2     = (const float*)d_in[36];
  const float* qk_bgate   = (const float*)d_in[37];
  const float* qk_bsca    = (const float*)d_in[38];

  char* ws = (char*)d_ws;
  float*  xq_s   = (float*) (ws + 0);
  float*  xk_s   = (float*) (ws + 262144);
  float*  xc_s   = (float*) (ws + 524288);
  float*  xq_v   = (float*) (ws + 786432);
  float*  xk_v   = (float*) (ws + 1179648);
  float*  xc_v   = (float*) (ws + 1572864);
  float*  ls     = (float*) (ws + 1966080);
  float*  lv     = (float*) (ws + 6160384);
  float*  ac     = (float*) (ws + 10354688);
  float*  sc     = (float*) (ws + 14548992);
  float*  vc     = (float*) (ws + 14811136);
  float*  atc    = (float*) (ws + 15204352);
  __bf16* Wv1b   = (__bf16*)(ws + 15253504);
  __bf16* Wscab  = (__bf16*)(ws + 15271936);
  __bf16* Wv2b   = (__bf16*)(ws + 15345664);
  __bf16* Wgateb = (__bf16*)(ws + 15357952);
  __bf16* W1p    = (__bf16*)(ws + 15374336);
  float*  b1p    = (float*) (ws + 15376384);
  float*  W2p    = (float*) (ws + 15376512);

  prep_kernel<<<dim3(241), dim3(256), 0, stream>>>(
      eb_Wv1, eb_Wsca, eb_Wv2, eb_Wgate, att_W1, att_b1, att_W2,
      Wv1b, Wscab, Wv2b, Wgateb, W1p, b1p, W2p);

  qkv_kernel<<<dim3(512), dim3(256), 0, stream>>>(
      x_sca, x_vec, qk_Wgate, qk_Wsca, qk_Wv1, qk_Wv2, qk_bgate, qk_bsca,
      xq_s, xk_s, xc_s, xq_v, xk_v, xc_v);

  edge_kernel<<<dim3(8192), dim3(128), 0, stream>>>(
      edge_sca, edge_vec, mask, coor, W_vecexp, dist_scale, eb_bsca, eb_bgate,
      Wv1b, Wscab, Wv2b, Wgateb, W1p, b1p, W2p, att_b2,
      xq_s, xk_s, xq_v, xk_v, ls, lv, ac);

  agg_kernel<<<dim3(512), dim3(256), 0, stream>>>(
      ls, lv, ac, xc_s, xc_v, coor, cn_scale, sc, vc, atc);

  final_kernel<<<dim3(512), dim3(32), 0, stream>>>(
      sc, vc, atc, coor, cp_Wact,
      cp_Wgate, cp_Wsca, cp_Wv1, cp_Wv2, cp_bgate, cp_bsca,
      cl_Wgate, cl_Wsca, cl_Wv1, cl_Wv2, cl_bgate, cl_bsca,
      combine, (float*)d_out);
}
